// MambaEncoder_7722351199035
// MI455X (gfx1250) — matrix operand order repaired and checked by reading, not compiled, not run on hardware
//
#include <hip/hip_runtime.h>

// ---------------------------------------------------------------------------
// MambaEncoder forward for MI455X (gfx1250, wave32, WMMA).
// GEMMs: bf16 v_wmma_f32_16x16x32_bf16, f32 accumulate, K-unrolled x2, with
// async global->LDS staging (global_load_async_to_lds_b128 + s_wait_asynccnt)
// on the aligned fast path. Everything else is f32 VALU. Scratch is a bump
// arena in d_ws (peak ~230 MB, reset per stage).
// ---------------------------------------------------------------------------

typedef __attribute__((ext_vector_type(16))) __bf16 v16bf;
typedef __attribute__((ext_vector_type(8)))  float  v8f;

union FragAB {
    v16bf v;
    uint4 q[2];
    unsigned short s[16];
};
union FragC {
    v8f v;
    float f[8];
};

static __device__ __forceinline__ unsigned short f2bf(float x) {
    unsigned int u = __float_as_uint(x);
    u = (u + 0x7fffu + ((u >> 16) & 1u)) >> 16;   // round-to-nearest-even
    return (unsigned short)u;
}

// Async DMA: copy 16B/lane from global to LDS (ASYNCcnt-tracked, CDNA5).
// LDS operand is the raw LDS byte offset = low 32 bits of the generic address
// (flat->LDS aperture).
static __device__ __forceinline__ void async_copy_b128(void* lds_dst,
                                                       const void* gsrc) {
    unsigned loff = (unsigned)(size_t)lds_dst;
    asm volatile("global_load_async_to_lds_b128 %0, %1, off"
                 :: "v"(loff), "v"(gsrc)
                 : "memory");
}

static __device__ __forceinline__ void wait_async_zero() {
#if __has_builtin(__builtin_amdgcn_s_wait_asynccnt)
    __builtin_amdgcn_s_wait_asynccnt(0);
#else
    asm volatile("s_wait_asynccnt 0x0" ::: "memory");
#endif
}

// ---------------------------------------------------------------------------
// Generic GEMM: C[M,N] = act( A[M,K](bf16) * W[N,K]^T(bf16) + bias ) (+= C)
// 256 threads = 8 waves; each wave owns a 16x16 C tile (block tile 128x16).
// K tile = 64 -> two WMMAs per iteration. Row stride 72 shorts keeps fragment
// ds_load_b128 16B-aligned and bank-conflict-free (r*36 mod 64 distinct).
// Out-of-range M/N are clamped (their C never stored); only ragged K zeros.
// act: 0 = none, 1 = softplus.  accum: 1 -> C += result.
// ---------------------------------------------------------------------------
#define KT   64
#define ASTR 72
#define BSTR 72

__global__ __launch_bounds__(256) void gemm_bf16_kernel(
    const unsigned short* __restrict__ A, int lda,
    const unsigned short* __restrict__ W, int ldw,
    float* __restrict__ C, int ldc,
    const float* __restrict__ bias,
    int M, int N, int K, int act, int accum)
{
    __shared__ __align__(16) unsigned short As[128 * ASTR];
    __shared__ __align__(16) unsigned short Bs[16 * BSTR];

    const int tid  = threadIdx.x;
    const int wv   = tid >> 5;
    const int lane = tid & 31;
    const int hf   = lane >> 4;       // half-wave select
    const int l16  = lane & 15;
    const int m0   = blockIdx.y * 128;
    const int n0   = blockIdx.x * 16;

    FragC acc;
#pragma unroll
    for (int i = 0; i < 8; ++i) acc.f[i] = 0.0f;

    for (int k0 = 0; k0 < K; k0 += KT) {
        const bool fastA = ((lda & 7) == 0) && (k0 + KT <= K);
        const bool fastB = ((ldw & 7) == 0) && (k0 + KT <= K);
        bool used_async = false;

        // ---- stage A tile (128 x 64) ----
        if (fastA) {
            // 1024 x 16B vectors, 4 per thread, rows clamped to M-1.
#pragma unroll
            for (int i = 0; i < 4; ++i) {
                int u  = tid * 4 + i;
                int r  = u >> 3;              // 8 vectors per row
                int cv = u & 7;
                int m  = m0 + r; if (m >= M) m = M - 1;
                async_copy_b128(&As[r * ASTR + cv * 8],
                                &A[(long)m * lda + k0 + cv * 8]);
            }
            used_async = true;
            // prefetch next K panel while DMA runs
            if (k0 + KT < K) {
                int pr = m0 + (tid >> 1);
                if (pr < M) __builtin_prefetch(&A[(long)pr * lda + k0 + KT], 0, 1);
            }
        } else {
            for (int idx = tid; idx < 128 * KT; idx += 256) {
                int r = idx >> 6, c = idx & 63;
                int m = m0 + r; if (m >= M) m = M - 1;
                int k = k0 + c;
                As[r * ASTR + c] = (k < K) ? A[(long)m * lda + k]
                                           : (unsigned short)0;
            }
        }

        // ---- stage B tile (16 x 64), stored as Bs[n][k] ----
        if (fastB) {
            if (tid < 128) {
                int r  = tid >> 3;            // n 0..15
                int cv = tid & 7;
                int nn = n0 + r; if (nn >= N) nn = N - 1;
                async_copy_b128(&Bs[r * BSTR + cv * 8],
                                &W[(long)nn * ldw + k0 + cv * 8]);
            }
            used_async = true;
        } else {
            for (int idx = tid; idx < 16 * KT; idx += 256) {
                int r = idx >> 6, c = idx & 63;
                int nn = n0 + r; if (nn >= N) nn = N - 1;
                int k = k0 + c;
                Bs[r * BSTR + c] = (k < K) ? W[(long)nn * ldw + k]
                                           : (unsigned short)0;
            }
        }

        if (used_async) wait_async_zero();
        __syncthreads();

        // ---- fragments (ISA 7.12.2 layouts) + 2x WMMA ----
        FragAB a0, a1, b0, b1;
        const unsigned short* ap = &As[(wv * 16 + l16) * ASTR + hf * 8];
        a0.q[0] = *(const uint4*)(ap);
        a0.q[1] = *(const uint4*)(ap + 16);
        a1.q[0] = *(const uint4*)(ap + 32);
        a1.q[1] = *(const uint4*)(ap + 48);
        const unsigned short* bp = &Bs[l16 * BSTR + hf * 8];
        b0.q[0] = *(const uint4*)(bp);
        b0.q[1] = *(const uint4*)(bp + 16);
        b1.q[0] = *(const uint4*)(bp + 32);
        b1.q[1] = *(const uint4*)(bp + 48);

        acc.v = __builtin_amdgcn_wmma_f32_16x16x32_bf16(
            false, a0.v, false, b0.v, (short)0, acc.v, false, false);
        acc.v = __builtin_amdgcn_wmma_f32_16x16x32_bf16(
            false, a1.v, false, b1.v, (short)0, acc.v, false, false);
        __syncthreads();
    }

    const int n = n0 + l16;
    if (n < N) {
        const float bv = bias ? bias[n] : 0.0f;
#pragma unroll
        for (int r = 0; r < 8; ++r) {
            int m = m0 + wv * 16 + r + hf * 8;   // C/D layout: row = r + 8*half
            if (m < M) {
                float v = acc.f[r] + bv;
                if (act == 1) v = (v > 20.0f) ? v : log1pf(expf(v));
                long ci = (long)m * ldc + n;
                if (accum) C[ci] += v; else C[ci] = v;
            }
        }
    }
}

// ---------------------------------------------------------------------------
// Patch-embed conv2d (direct), writes token-major (row, c) layout.
// ---------------------------------------------------------------------------
__global__ void conv2d_kernel(const float* __restrict__ in,
                              const float* __restrict__ w,
                              const float* __restrict__ bias,
                              float* __restrict__ tok,
                              int Bn, int Cin, int Hin, int Win,
                              int Cout, int Hout, int Wout, int P, int S, int pad)
{
    long t = (long)blockIdx.x * blockDim.x + threadIdx.x;
    long HWo = (long)Hout * Wout;
    long total = (long)Bn * HWo * Cout;
    if (t >= total) return;
    int  c  = (int)(t % Cout);
    long rs = t / Cout;
    int  s  = (int)(rs % HWo);
    int  n  = (int)(rs / HWo);
    int  oh = s / Wout, ow = s % Wout;

    float acc = bias[c];
    for (int ic = 0; ic < Cin; ++ic)
        for (int kh = 0; kh < P; ++kh) {
            int ih = oh * S - pad + kh;
            if (ih < 0 || ih >= Hin) continue;
            for (int kw = 0; kw < P; ++kw) {
                int iw = ow * S - pad + kw;
                if (iw < 0 || iw >= Win) continue;
                acc += in[((long)(n * Cin + ic) * Hin + ih) * Win + iw] *
                       w[((c * Cin + ic) * P + kh) * P + kw];
            }
        }
    tok[t] = acc;   // t == (n*HWo + s)*Cout + c
}

// ---------------------------------------------------------------------------
// LayerNorm per token row. Optional f32 and/or bf16 outputs.
// ---------------------------------------------------------------------------
__global__ void ln_kernel(const float* __restrict__ src,
                          float* __restrict__ dstF,
                          unsigned short* __restrict__ dstB,
                          const float* __restrict__ w,
                          const float* __restrict__ b,
                          long rows, int C)
{
    long r = (long)blockIdx.x * blockDim.x + threadIdx.x;
    if (r >= rows) return;
    const float* p = src + r * C;
    float m = 0.0f;
    for (int i = 0; i < C; ++i) m += p[i];
    m /= (float)C;
    float v = 0.0f;
    for (int i = 0; i < C; ++i) { float d = p[i] - m; v += d * d; }
    v /= (float)C;
    float inv = rsqrtf(v + 1e-5f);
    for (int i = 0; i < C; ++i) {
        float o = (p[i] - m) * inv * w[i] + b[i];
        if (dstF) dstF[r * C + i] = o;
        if (dstB) dstB[r * C + i] = f2bf(o);
    }
}

// f32 -> bf16 (optionally strided source rows).
__global__ void cvt_bf16_kernel(const float* __restrict__ src, long srcStride,
                                int cols, long total,
                                unsigned short* __restrict__ dst)
{
    long t = (long)blockIdx.x * blockDim.x + threadIdx.x;
    if (t >= total) return;
    long r = t / cols;
    int  c = (int)(t % cols);
    dst[t] = f2bf(src[r * srcStride + c]);
}

// Causal depthwise conv1d (k=4, left pad 3) + bias + SiLU. Input = xs slice of
// xz (stride 2*di). Writes f32 + bf16.
__global__ void dwconv1d_silu_kernel(const float* __restrict__ xz, int twoDi, int di,
                                     const float* __restrict__ cw,
                                     const float* __restrict__ cb,
                                     float* __restrict__ xs,
                                     unsigned short* __restrict__ xsb,
                                     long L, long total)
{
    long t = (long)blockIdx.x * blockDim.x + threadIdx.x;
    if (t >= total) return;
    int  c   = (int)(t % di);
    long row = t / di;
    long l   = row % L;
    float acc = cb[c];
#pragma unroll
    for (int j = 0; j < 4; ++j) {
        long l2 = l - 3 + j;
        if (l2 >= 0) acc += xz[(row - 3 + j) * twoDi + c] * cw[c * 4 + j];
    }
    float s = acc / (1.0f + expf(-acc));
    xs[t]  = s;
    xsb[t] = f2bf(s);
}

// Selective scan: one lane per (b, channel, state n); 16-lane shfl reduction
// for einsum('bdn,bn->bd', h, C).
__global__ void scan_kernel(const float* __restrict__ dt,
                            const float* __restrict__ xs,
                            const float* __restrict__ proj, int nproj, int dtr,
                            const float* __restrict__ A_log,
                            float* __restrict__ ys,
                            int B, long L, int di)
{
    int t  = blockIdx.x * blockDim.x + threadIdx.x;
    int n  = t & 15;
    int cc = t >> 4;
    int c  = cc % di;
    int bb = cc / di;
    if (bb >= B) return;
    float a = -expf(A_log[c * 16 + n]);
    float h = 0.0f;
    for (long step = 0; step < L; ++step) {
        long  row = (long)bb * L + step;
        float dtv = dt[row * di + c];
        float xv  = xs[row * di + c];
        float Bv  = proj[row * nproj + dtr + n];
        float Cv  = proj[row * nproj + dtr + 16 + n];
        h = expf(dtv * a) * h + dtv * xv * Bv;
        float p = h * Cv;
        p += __shfl_xor(p, 1, 32);
        p += __shfl_xor(p, 2, 32);
        p += __shfl_xor(p, 4, 32);
        p += __shfl_xor(p, 8, 32);
        if (n == 0) ys[row * di + c] = p;
    }
}

// y = (ys + xs*D) * silu(z);  z = xz[:, di:2di]. Writes f32 (in place) + bf16.
__global__ void ygate_kernel(float* __restrict__ y,
                             const float* __restrict__ xs,
                             const float* __restrict__ Dp,
                             const float* __restrict__ xz, int twoDi, int di,
                             long total, unsigned short* __restrict__ ybf)
{
    long t = (long)blockIdx.x * blockDim.x + threadIdx.x;
    if (t >= total) return;
    int  c   = (int)(t % di);
    long row = t / di;
    float v = y[t] + xs[t] * Dp[c];
    float z = xz[row * twoDi + di + c];
    v *= z / (1.0f + expf(-z));
    y[t]   = v;
    ybf[t] = f2bf(v);
}

// Depthwise 3x3x3 conv over (nf,H,W) + bias + exact GELU; emits bf16 only.
__global__ void dwconv3d_gelu_kernel(const float* __restrict__ h1,
                                     const float* __restrict__ w,
                                     const float* __restrict__ bias,
                                     unsigned short* __restrict__ out_bf,
                                     int NFv, int H, int W, int C, long total)
{
    long t = (long)blockIdx.x * blockDim.x + threadIdx.x;
    if (t >= total) return;
    long HW = (long)H * W;
    long L  = (long)NFv * HW;
    int  c   = (int)(t % C);
    long row = t / C;
    long l   = row % L;
    long bb  = row / L;
    int  f   = (int)(l / HW);
    long s   = l % HW;
    int  yy  = (int)(s / W), xx = (int)(s % W);

    float acc = bias[c];
    for (int kd = 0; kd < 3; ++kd) {
        int f2 = f + kd - 1;
        if (f2 < 0 || f2 >= NFv) continue;
        for (int kh = 0; kh < 3; ++kh) {
            int y2 = yy + kh - 1;
            if (y2 < 0 || y2 >= H) continue;
            for (int kw = 0; kw < 3; ++kw) {
                int x2 = xx + kw - 1;
                if (x2 < 0 || x2 >= W) continue;
                long src = (bb * L + (long)f2 * HW + (long)y2 * W + x2) * C + c;
                acc += h1[src] * w[c * 27 + kd * 9 + kh * 3 + kw];
            }
        }
    }
    float g = 0.5f * acc * (1.0f + erff(acc * 0.70710678118f));
    out_bf[t] = f2bf(g);
}

// Token-major (row, c) -> NCHW stage output (coalesced writes).
__global__ void tok2nchw_kernel(const float* __restrict__ xf,
                                float* __restrict__ out,
                                int C, long HW, long total)
{
    long t = (long)blockIdx.x * blockDim.x + threadIdx.x;
    if (t >= total) return;
    long s   = t % HW;
    int  c   = (int)((t / HW) % C);
    long nfi = t / (HW * C);
    out[t] = xf[(nfi * HW + s) * C + c];
}

// ---------------------------------------------------------------------------
// Host orchestration
// ---------------------------------------------------------------------------
extern "C" void kernel_launch(void* const* d_in, const int* in_sizes, int n_in,
                              void* d_out, int out_size, void* d_ws, size_t ws_size,
                              hipStream_t stream)
{
    (void)in_sizes; (void)n_in; (void)out_size; (void)ws_size;

    static const int DIMS_[4] = {64, 128, 256, 512};
    static const int INCH_[4] = {2, 64, 128, 256};
    static const int PS_[4]   = {7, 3, 3, 3};
    static const int ST_[4]   = {4, 2, 2, 2};
    const int BZ = 4, NFv = 5;
    const int BN = BZ * NFv;   // 20

    int cur = 0;
    auto F = [&]() -> const float* { return (const float*)d_in[cur++]; };

    const float* X = F();   // (4,5,2,224,224) == (20,2,224,224)

    char* wsBase = (char*)d_ws;
    float* outF  = (float*)d_out;

    auto NB = [](long n, int b) -> unsigned { return (unsigned)((n + b - 1) / b); };
    const int TB = 256;

    const float* cur_in = X;
    int Hin = 224, Win = 224;
    size_t outOff = 0;

    for (int st = 0; st < 4; ++st) {
        const int Cin = INCH_[st], d = DIMS_[st], P = PS_[st], S = ST_[st];
        const int pad = P / 2;
        const int H = (Hin + 2 * pad - P) / S + 1;
        const int W = (Win + 2 * pad - P) / S + 1;
        const long HW = (long)H * W;
        const long L  = (long)NFv * HW;
        const long M  = (long)BZ * L;       // token rows (== 20*HW)
        const int di  = 2 * d;
        const int hid = 4 * d;
        const int dtr = d / 16;
        const int nproj = dtr + 2 * 16;

        // ---- stage arena ----
        size_t wsOff = 0;
        auto alloc = [&](size_t bytes) -> void* {
            wsOff = (wsOff + 255) & ~(size_t)255;
            void* p = wsBase + wsOff;
            wsOff += bytes;
            return p;
        };
        float*          xf    = (float*)alloc((size_t)M * d  * 4);
        unsigned short* lnbf  = (unsigned short*)alloc((size_t)M * d  * 2);
        float*          xz    = (float*)alloc((size_t)M * 2 * di * 4);   // also fc1 out
        float*          xsc   = (float*)alloc((size_t)M * di * 4);
        unsigned short* xscb  = (unsigned short*)alloc((size_t)M * di * 2);
        float*          proj  = (float*)alloc((size_t)M * nproj * 4);
        unsigned short* dtinb = (unsigned short*)alloc((size_t)M * dtr * 2);
        float*          dtb   = (float*)alloc((size_t)M * di * 4);        // dt; later h2 bf16
        float*          ybuf  = (float*)alloc((size_t)M * di * 4);
        unsigned short* ybf   = (unsigned short*)alloc((size_t)M * di * 2);
        unsigned short* wbuf  = (unsigned short*)alloc((size_t)4 * d * d * 2);
        float*          h1    = xz;                           // alias (same size)
        unsigned short* h2b   = (unsigned short*)dtb;         // alias (same bytes)

        // ---- stage params ----
        const float* proj_w  = F();
        const float* proj_b  = F();
        const float* pe_ln_w = F();
        const float* pe_ln_b = F();
        const float* ln_w    = F();
        const float* ln_b    = F();

        // ---- patch embed + two LNs ----
        conv2d_kernel<<<NB(M * d, TB), TB, 0, stream>>>(
            cur_in, proj_w, proj_b, xf, BN, Cin, Hin, Win, d, H, W, P, S, pad);
        ln_kernel<<<NB(M, TB), TB, 0, stream>>>(xf, xf, nullptr, pe_ln_w, pe_ln_b, M, d);
        ln_kernel<<<NB(M, TB), TB, 0, stream>>>(xf, xf, nullptr, ln_w, ln_b, M, d);

        // ---- block params (depth 1) ----
        const float* n1w = F(); const float* n1b = F();
        const float* n2w = F(); const float* n2b = F();
        const float* in_proj_w = F();
        const float* conv_w = F(); const float* conv_b = F();
        const float* x_proj_w = F();
        const float* dt_proj_w = F(); const float* dt_proj_b = F();
        const float* A_log = F(); const float* Dp = F();
        const float* out_proj_w = F();
        const float* fc1_w = F(); const float* fc1_b = F();
        const float* dw_w = F();  const float* dw_b = F();
        const float* fc2_w = F(); const float* fc2_b = F();

        dim3 blk(TB);

        // -------- Mamba mixer --------
        // ln1 -> bf16
        ln_kernel<<<NB(M, TB), blk, 0, stream>>>(xf, nullptr, lnbf, n1w, n1b, M, d);

        // xz = ln1 @ in_proj_w^T   (M x 2di)
        cvt_bf16_kernel<<<NB((long)2 * di * d, TB), blk, 0, stream>>>(
            in_proj_w, d, d, (long)2 * di * d, wbuf);
        {
            dim3 g((2 * di + 15) / 16, (unsigned)((M + 127) / 128));
            gemm_bf16_kernel<<<g, blk, 0, stream>>>(lnbf, d, wbuf, d, xz, 2 * di,
                                                    nullptr, (int)M, 2 * di, d, 0, 0);
        }

        // causal dwconv1d + SiLU on xs slice
        dwconv1d_silu_kernel<<<NB(M * di, TB), blk, 0, stream>>>(
            xz, 2 * di, di, conv_w, conv_b, xsc, xscb, L, M * di);

        // proj = xs @ x_proj_w^T   (M x nproj)
        cvt_bf16_kernel<<<NB((long)nproj * di, TB), blk, 0, stream>>>(
            x_proj_w, di, di, (long)nproj * di, wbuf);
        {
            dim3 g((nproj + 15) / 16, (unsigned)((M + 127) / 128));
            gemm_bf16_kernel<<<g, blk, 0, stream>>>(xscb, di, wbuf, di, proj, nproj,
                                                    nullptr, (int)M, nproj, di, 0, 0);
        }

        // dt = softplus(proj[:, :dtr] @ dt_proj_w^T + b)   (M x di)
        cvt_bf16_kernel<<<NB(M * dtr, TB), blk, 0, stream>>>(
            proj, nproj, dtr, M * dtr, dtinb);
        cvt_bf16_kernel<<<NB((long)di * dtr, TB), blk, 0, stream>>>(
            dt_proj_w, dtr, dtr, (long)di * dtr, wbuf);
        {
            dim3 g((di + 15) / 16, (unsigned)((M + 127) / 128));
            gemm_bf16_kernel<<<g, blk, 0, stream>>>(dtinb, dtr, wbuf, dtr, dtb, di,
                                                    dt_proj_b, (int)M, di, dtr, 1, 0);
        }

        // selective scan -> ys
        scan_kernel<<<NB((long)BZ * di * 16, TB), blk, 0, stream>>>(
            dtb, xsc, proj, nproj, dtr, A_log, ybuf, BZ, L, di);

        // y = (ys + xs*D) * silu(z)  -> bf16
        ygate_kernel<<<NB(M * di, TB), blk, 0, stream>>>(
            ybuf, xsc, Dp, xz, 2 * di, di, M * di, ybf);

        // xf += y @ out_proj_w^T
        cvt_bf16_kernel<<<NB((long)d * di, TB), blk, 0, stream>>>(
            out_proj_w, di, di, (long)d * di, wbuf);
        {
            dim3 g((d + 15) / 16, (unsigned)((M + 127) / 128));
            gemm_bf16_kernel<<<g, blk, 0, stream>>>(ybf, di, wbuf, di, xf, d,
                                                    nullptr, (int)M, d, di, 0, 1);
        }

        // -------- MLP with dwconv3d --------
        ln_kernel<<<NB(M, TB), blk, 0, stream>>>(xf, nullptr, lnbf, n2w, n2b, M, d);

        // h1 = ln2 @ fc1_w^T + b1   (M x hid)   [h1 aliases xz]
        cvt_bf16_kernel<<<NB((long)hid * d, TB), blk, 0, stream>>>(
            fc1_w, d, d, (long)hid * d, wbuf);
        {
            dim3 g((hid + 15) / 16, (unsigned)((M + 127) / 128));
            gemm_bf16_kernel<<<g, blk, 0, stream>>>(lnbf, d, wbuf, d, h1, hid,
                                                    fc1_b, (int)M, hid, d, 0, 0);
        }

        // dwconv3d + bias + GELU -> bf16   [h2b aliases dt]
        dwconv3d_gelu_kernel<<<NB(M * hid, TB), blk, 0, stream>>>(
            h1, dw_w, dw_b, h2b, NFv, H, W, hid, M * hid);

        // xf += gelu(...) @ fc2_w^T + b2
        cvt_bf16_kernel<<<NB((long)d * hid, TB), blk, 0, stream>>>(
            fc2_w, hid, hid, (long)d * hid, wbuf);
        {
            dim3 g((d + 15) / 16, (unsigned)((M + 127) / 128));
            gemm_bf16_kernel<<<g, blk, 0, stream>>>(h2b, hid, wbuf, hid, xf, d,
                                                    fc2_b, (int)M, d, hid, 0, 1);
        }

        // ---- stage output (also next stage's conv input) ----
        long stageElems = (long)BN * d * HW;
        tok2nchw_kernel<<<NB(stageElems, TB), blk, 0, stream>>>(
            xf, outF + outOff, d, HW, stageElems);

        cur_in = outF + outOff;
        outOff += (size_t)stageElems;
        Hin = H; Win = W;
    }
}
